// AttentionSampling_12292196402040
// MI455X (gfx1250) — matrix operand order repaired and checked by reading, not compiled, not run on hardware
//
#include <hip/hip_runtime.h>

// Problem constants (from reference setup_inputs)
#define B_    4
#define SQ    2048
#define SKV   8192
#define D_    1024
#define QROWS 16          // q rows per workgroup tile
#define KROWS 64          // kv rows per workgroup tile (QROWS * factor)
#define ROWPAD 1032       // bf16 elements per padded LDS row (1024 + 8 -> conflict-free b128 frag loads)
#define LN_EPS 1e-5f
#define NTHREADS 512      // 16 waves: LDS limits us to 1 WG/WGP, so pack waves for HBM latency hiding
#define NWAVES  16

typedef __attribute__((ext_vector_type(16))) __bf16 v16bf;
typedef __attribute__((ext_vector_type(8)))  float  v8f;

union Frag16 {
    v16bf v;
    uint4 q[2];
};

__device__ __forceinline__ unsigned short f2bf(float f) {
    // round-to-nearest-even f32 -> bf16
    unsigned u = __float_as_uint(f);
    u += 0x7FFFu + ((u >> 16) & 1u);
    return (unsigned short)(u >> 16);
}

__device__ __forceinline__ float waveSum(float x) {
    #pragma unroll
    for (int off = 16; off > 0; off >>= 1) x += __shfl_xor(x, off, 32);
    return x;
}

// LDS layout (dynamic):
//   s_w   : D_ floats                (ln weight)
//   s_b   : D_ floats                (ln bias)
//   s_qn  : QROWS*ROWPAD bf16        (normalized q, bf16)
//   s_kn  : KROWS*ROWPAD bf16        (normalized k, bf16)
//   s_wf  : 4*4*16*16 floats         (raw 16x16 score tiles: [kquarter][ktile][M][N])
#define SHMEM_BYTES ((2*D_)*4 + (QROWS+KROWS)*ROWPAD*2 + 4*4*16*16*4)

__global__ void __launch_bounds__(NTHREADS, 1)
attn_downsample_kernel(const float* __restrict__ qg,
                       const float* __restrict__ kg,
                       const float* __restrict__ vg,
                       const float* __restrict__ lnw,
                       const float* __restrict__ lnb,
                       float* __restrict__ outg)
{
    extern __shared__ char smem[];
    float*          s_w  = (float*)smem;                               // D_
    float*          s_b  = s_w + D_;                                   // D_
    unsigned short* s_qn = (unsigned short*)(s_b + D_);                // QROWS*ROWPAD
    unsigned short* s_kn = s_qn + QROWS * ROWPAD;                      // KROWS*ROWPAD
    float*          s_wf = (float*)(s_kn + KROWS * ROWPAD);            // 4*4*16*16

    const int tid  = threadIdx.x;
    const int lane = tid & 31;
    const int wave = tid >> 5;

    const int blk  = blockIdx.x;
    const int b    = blk >> 7;          // / (SQ/QROWS) = /128
    const int tile = blk & 127;
    const int s0   = tile * QROWS;      // first q row of tile
    const int r0   = tile * KROWS;      // first kv row of tile

    // ---- stage 0: LN weight/bias into LDS ----
    for (int i = tid; i < D_; i += NTHREADS) { s_w[i] = lnw[i]; s_b[i] = lnb[i]; }
    __syncthreads();

    // ---- pass 1: LayerNorm q (16 rows) + k (64 rows) -> bf16 in LDS ----
    // 80 rows total, wave w handles rows w, w+16, ..., w+64 (5 rows).
    // Double-buffered: next row's 8 global b128 loads are issued before the
    // current row's cross-lane reduction, keeping ~16 loads in flight/wave.
    {
        const float4* sp = (wave < QROWS)
            ? (const float4*)(qg + ((size_t)(b * SQ + s0 + wave)) * D_)
            : (const float4*)(kg + ((size_t)(b * SKV + r0 + (wave - QROWS))) * D_);
        float4 x[8];
        #pragma unroll
        for (int c = 0; c < 8; ++c) x[c] = sp[c * 32 + lane];   // d = c*128 + lane*4

        #pragma unroll 1
        for (int j = 0; j < 5; ++j) {
            const int row = wave + NWAVES * j;

            // prefetch next row
            float4 xn[8];
            if (j < 4) {
                const int nrow = row + NWAVES;
                const float4* np = (nrow < QROWS)
                    ? (const float4*)(qg + ((size_t)(b * SQ + s0 + nrow)) * D_)
                    : (const float4*)(kg + ((size_t)(b * SKV + r0 + (nrow - QROWS))) * D_);
                #pragma unroll
                for (int c = 0; c < 8; ++c) xn[c] = np[c * 32 + lane];
            }

            float sum = 0.f, ssq = 0.f;
            #pragma unroll
            for (int c = 0; c < 8; ++c) {
                sum += x[c].x + x[c].y + x[c].z + x[c].w;
                ssq += x[c].x * x[c].x + x[c].y * x[c].y + x[c].z * x[c].z + x[c].w * x[c].w;
            }
            sum = waveSum(sum);
            ssq = waveSum(ssq);
            const float mu   = sum * (1.0f / D_);
            const float rstd = rsqrtf(ssq * (1.0f / D_) - mu * mu + LN_EPS);

            unsigned short* dst = (row < QROWS) ? (s_qn + row * ROWPAD)
                                                : (s_kn + (row - QROWS) * ROWPAD);
            #pragma unroll
            for (int c = 0; c < 8; ++c) {
                const int d = c * 128 + lane * 4;
                const float y0 = (x[c].x - mu) * rstd * s_w[d + 0] + s_b[d + 0];
                const float y1 = (x[c].y - mu) * rstd * s_w[d + 1] + s_b[d + 1];
                const float y2 = (x[c].z - mu) * rstd * s_w[d + 2] + s_b[d + 2];
                const float y3 = (x[c].w - mu) * rstd * s_w[d + 3] + s_b[d + 3];
                uint2 pk;
                pk.x = (unsigned)f2bf(y0) | ((unsigned)f2bf(y1) << 16);
                pk.y = (unsigned)f2bf(y2) | ((unsigned)f2bf(y3) << 16);
                *(uint2*)(dst + d) = pk;
            }

            if (j < 4) {
                #pragma unroll
                for (int c = 0; c < 8; ++c) x[c] = xn[c];
            }
        }
    }
    __syncthreads();

    // ---- pass 2: block QK^T via WMMA bf16 ----
    // wave = h*4 + t : key-tile t (keys 16t..16t+15), K-quarter h (d in [h*256, h*256+256))
    {
        const int t = wave & 3;
        const int h = wave >> 2;
        v8f c = {};
        const int arow = lane & 15;
        const int aSel = (lane < 16) ? 0 : 8;    // A 16x32 bf16: lanes 0-15 -> K 0..7 & 16..23; 16-31 -> 8..15 & 24..31
        const int bSel = (lane < 16) ? 0 : 8;    // B 32x16 bf16: lanes 0-15 -> K 0..7 & 16..23; 16-31 -> 8..15 & 24..31

        #pragma unroll
        for (int step = 0; step < 8; ++step) {
            const int kb = h * 256 + step * 32;
            Frag16 af, bf;
            const unsigned short* ap = s_qn + arow * ROWPAD + kb + aSel;
            af.q[0] = *(const uint4*)(ap);        // 8 bf16
            af.q[1] = *(const uint4*)(ap + 16);   // 8 bf16
            const unsigned short* bp = s_kn + (t * 16 + arow) * ROWPAD + kb + bSel;
            bf.q[0] = *(const uint4*)(bp);
            bf.q[1] = *(const uint4*)(bp + 16);
            c = __builtin_amdgcn_wmma_f32_16x16x32_bf16(
                    false, af.v, false, bf.v, (short)0, c, false, false);
        }
        // C layout: VGPR j -> M = j + (lane<16?0:8), N = lane&15
        #pragma unroll
        for (int j = 0; j < 8; ++j) {
            const int M = j + ((lane < 16) ? 0 : 8);
            const int N = lane & 15;
            s_wf[(((h * 4 + t) * 16 + M) * 16) + N] = c[j];
        }
    }
    __syncthreads();

    // ---- pass 3: LayerNorm v rows on the fly, weighted sum + residual ----
    // wave owns output row m = wave; v rows double-buffered like pass 1.
    {
        const int m  = wave;
        const int tt = m >> 2;                    // key-tile containing keys 4m..4m+3
        float4 acc[8];
        #pragma unroll
        for (int c = 0; c < 8; ++c) acc[c] = make_float4(0.f, 0.f, 0.f, 0.f);

        const float4* vp0 = (const float4*)(vg + ((size_t)(b * SKV + r0 + 4 * m)) * D_);
        float4 x[8];
        #pragma unroll
        for (int c = 0; c < 8; ++c) x[c] = vp0[c * 32 + lane];

        #pragma unroll 1
        for (int f = 0; f < 4; ++f) {
            // prefetch next v row
            float4 xn[8];
            if (f < 3) {
                const float4* np = (const float4*)(vg + ((size_t)(b * SKV + r0 + 4 * m + f + 1)) * D_);
                #pragma unroll
                for (int c = 0; c < 8; ++c) xn[c] = np[c * 32 + lane];
            }

            const int n = 4 * (m & 3) + f;        // column inside the tile
            float wc = 0.f;
            #pragma unroll
            for (int h = 0; h < 4; ++h)
                wc += s_wf[(((h * 4 + tt) * 16 + m) * 16) + n];

            float sum = 0.f, ssq = 0.f;
            #pragma unroll
            for (int c = 0; c < 8; ++c) {
                sum += x[c].x + x[c].y + x[c].z + x[c].w;
                ssq += x[c].x * x[c].x + x[c].y * x[c].y + x[c].z * x[c].z + x[c].w * x[c].w;
            }
            sum = waveSum(sum);
            ssq = waveSum(ssq);
            const float mu   = sum * (1.0f / D_);
            const float rstd = rsqrtf(ssq * (1.0f / D_) - mu * mu + LN_EPS);

            #pragma unroll
            for (int c = 0; c < 8; ++c) {
                const int d = c * 128 + lane * 4;
                acc[c].x += wc * ((x[c].x - mu) * rstd * s_w[d + 0] + s_b[d + 0]);
                acc[c].y += wc * ((x[c].y - mu) * rstd * s_w[d + 1] + s_b[d + 1]);
                acc[c].z += wc * ((x[c].z - mu) * rstd * s_w[d + 2] + s_b[d + 2]);
                acc[c].w += wc * ((x[c].w - mu) * rstd * s_w[d + 3] + s_b[d + 3]);
            }

            if (f < 3) {
                #pragma unroll
                for (int c = 0; c < 8; ++c) x[c] = xn[c];
            }
        }

        // residual add with raw q (L2 hit) and store
        const float4* qp = (const float4*)(qg + ((size_t)(b * SQ + s0 + m)) * D_);
        float4*       op = (float4*)(outg + ((size_t)(b * SQ + s0 + m)) * D_);
        #pragma unroll
        for (int c = 0; c < 8; ++c) {
            const float4 qv = qp[c * 32 + lane];
            float4 o;
            o.x = qv.x + acc[c].x;
            o.y = qv.y + acc[c].y;
            o.z = qv.z + acc[c].z;
            o.w = qv.w + acc[c].w;
            op[c * 32 + lane] = o;
        }
    }
}

extern "C" void kernel_launch(void* const* d_in, const int* in_sizes, int n_in,
                              void* d_out, int out_size, void* d_ws, size_t ws_size,
                              hipStream_t stream) {
    const float* q   = (const float*)d_in[0];
    const float* k   = (const float*)d_in[1];
    const float* v   = (const float*)d_in[2];
    const float* lnw = (const float*)d_in[3];
    const float* lnb = (const float*)d_in[4];
    float* out = (float*)d_out;

    const size_t shmem = SHMEM_BYTES;   // ~185 KB of the 320 KB WGP LDS -> 1 WG per WGP, 16 waves
    (void)hipFuncSetAttribute((const void*)attn_downsample_kernel,
                              hipFuncAttributeMaxDynamicSharedMemorySize, (int)shmem);

    const int nBlocks = B_ * (SQ / QROWS);   // 4 * 128 = 512 workgroups
    attn_downsample_kernel<<<dim3(nBlocks), dim3(NTHREADS), shmem, stream>>>(q, k, v, lnw, lnb, out);
}
